// AFNO3DHydraLora_14259291423364
// MI455X (gfx1250) — compile-verified, matrix operand order checked by reading, not run on hardware
//
#include <hip/hip_runtime.h>
#include <math.h>

// MI455X / gfx1250, wave32. All matmuls via v_wmma_f32_16x16x32_f16.
typedef __attribute__((ext_vector_type(16))) _Float16 v16h;
typedef __attribute__((ext_vector_type(8)))  float    v8f;

#define WMMA(A_, B_, C_) __builtin_amdgcn_wmma_f32_16x16x32_f16(false,(A_),false,(B_),(short)0,(C_),false,false)

// Problem constants
// x: [4,64,64,32,128], kept modes [32,32,8], NB=8 blocks x BS=16 feats, R=32, E=4
#define SZ_S1 16777216   // 4*64*64*8*128
#define SZ_S2 8388608    // 4*64*32*8*128
#define SZ_S3 4194304    // 4*32*32*8*128

// 16-bit A/B per-lane K mapping (ISA 7.12.2): half h of lane l (hi8 = (l>>4)*8):
//   k = h<8 ? hi8+h : 16+hi8+(h-8)
__device__ __forceinline__ int kmap(int h, int hi8){ return (h < 8) ? (hi8 + h) : (16 + hi8 + (h - 8)); }

// ---------------- DFT coefficient tables (packed in A-operand register layout) --------------
// TAB ids: 0: T_L (1 mat) | 1..16: T_W (kwt*8+mat*4+ch) | 17..32: T_IW (mt*4+mat*2+ch) | 33..34: T_IL
__global__ void k_init_tables(_Float16* __restrict__ TAB){
  const float PI2 = 6.283185307179586f;
  const float s32 = 0.17677669529663687f;   // 1/sqrt(32)
  int id = blockIdx.x;
  int t  = threadIdx.x;
  int l = t >> 4, h = t & 15;
  int m = l & 15, hi8 = (l >> 4) * 8;
  int kl = kmap(h, hi8);
  float val = 0.f;
  if (id == 0) {
    // forward L rfft: out_r[t]=sum v cos, out_i[t]=-sum v sin ; theta=2pi t k/32
    if (m < 8) val =  cosf(PI2 * (float)(m * kl) / 32.f) * s32;
    else       val = -sinf(PI2 * (float)((m - 8) * kl) / 32.f) * s32;
  } else if (id <= 16) {
    // forward 64->32 DFT (W and H axes). Kcat=128: kk<64 real part, else imag.
    int idx = id - 1; int kwt = idx >> 3; int mat = (idx >> 2) & 1; int ch = idx & 3;
    int kk = ch * 32 + kl; int w = kk & 63; int part = kk >> 6;
    int kw = kwt * 16 + m;
    float th = PI2 * (float)(kw * w) / 64.f;
    float c = cosf(th) * 0.125f, s = sinf(th) * 0.125f;
    // out_r = sum vr*c + vi*s ; out_i = sum vi*c - vr*s
    val = (mat == 0) ? (part ? s : c) : (part ? c : -s);
  } else if (id <= 32) {
    // inverse 32->64 DFT. Kcat=64: kk<32 real, else imag.
    int idx = id - 17; int mt = idx >> 2; int mat = (idx >> 1) & 1; int ch = idx & 1;
    int kk = ch * 32 + kl; int k = kk & 31; int part = kk >> 5;
    int n = mt * 16 + m;
    float th = PI2 * (float)(n * k) / 64.f;
    float c = cosf(th) * 0.125f, s = sinf(th) * 0.125f;
    // out_r = Xr*c - Xi*s ; out_i = Xr*s + Xi*c
    val = (mat == 0) ? (part ? -s : c) : (part ? c : s);
  } else {
    // inverse L (c2r, bins 0..7 nonzero): y[n]=s32*(Fr0 + 2*sum_{k=1..7}(Frk cos - Fik sin))
    int mt = id - 33;
    int n = mt * 16 + m;
    if (kl < 8) {
      int k = kl;
      val = ((k == 0) ? 1.f : 2.f * cosf(PI2 * (float)(k * n) / 32.f)) * s32;
    } else if (kl < 16) {
      int k = kl - 8;
      val = (k == 0) ? 0.f : -2.f * sinf(PI2 * (float)(k * n) / 32.f) * s32;
    } else val = 0.f;
  }
  TAB[id * 512 + l * 16 + h] = (_Float16)val;
}

// ---------------- Effective MLP weights: Weff = W + (1/32) * A @ (sum_e ew_e B_e) ----------
__device__ float weff_entry(const float* __restrict__ w, const float* __restrict__ A,
                            const float* __restrict__ Bx, const float* __restrict__ ew,
                            int nb, int i, int o, int wsel){
  float e0 = ew[0], e1 = ew[1], e2 = ew[2], e3 = ew[3];
  float acc = 0.f;
  for (int r = 0; r < 32; r++) {
    float bm = e0 * Bx[((nb * 4 + 0) * 32 + r) * 16 + o] + e1 * Bx[((nb * 4 + 1) * 32 + r) * 16 + o]
             + e2 * Bx[((nb * 4 + 2) * 32 + r) * 16 + o] + e3 * Bx[((nb * 4 + 3) * 32 + r) * 16 + o];
    acc += A[(nb * 16 + i) * 32 + r] * bm;
  }
  return w[((wsel * 8 + nb) * 16 + i) * 16 + o] + (1.f / 32.f) * acc;
}

// Pack 4 concat-K A-operands per block nb:
// mat0=[W1r;-W1i] mat1=[W1i;W1r] mat2=[W2r;-W2i] mat3=[W2i;W2r]  (A[o][k] = Wcat[k][o])
__global__ void k_init_weff(const float* w1, const float* b1, const float* w2, const float* b2,
                            const float* A1r, const float* B1r, const float* A1i, const float* B1i,
                            const float* A2r, const float* B2r, const float* A2i, const float* B2i,
                            const float* ew, _Float16* __restrict__ WPK){
  __shared__ float Wm[4][16][16];
  int nb = blockIdx.x;
  int t = threadIdx.x;
  int i = t >> 4, o = t & 15;
  Wm[0][i][o] = weff_entry(w1, A1r, B1r, ew, nb, i, o, 0);
  Wm[1][i][o] = weff_entry(w1, A1i, B1i, ew, nb, i, o, 1);
  Wm[2][i][o] = weff_entry(w2, A2r, B2r, ew, nb, i, o, 0);
  Wm[3][i][o] = weff_entry(w2, A2i, B2i, ew, nb, i, o, 1);
  __syncthreads();
  int mat = t >> 6; int l = (t & 63) >> 1; int h0 = (t & 1) * 8;
  int oo = l & 15; int hi8 = (l >> 4) * 8;
  for (int h = h0; h < h0 + 8; h++) {
    int k = kmap(h, hi8);
    float v;
    if      (mat == 0) v = (k < 16) ?  Wm[0][k][oo] : -Wm[1][k - 16][oo];
    else if (mat == 1) v = (k < 16) ?  Wm[1][k][oo] :  Wm[0][k - 16][oo];
    else if (mat == 2) v = (k < 16) ?  Wm[2][k][oo] : -Wm[3][k - 16][oo];
    else               v = (k < 16) ?  Wm[3][k][oo] :  Wm[2][k - 16][oo];
    WPK[((nb * 4 + mat) * 32 + l) * 16 + h] = (_Float16)v;
  }
}

// ---------------- Stage 1: rfft along L=32, keep 8 modes. One WMMA per 16-c tile. ----------
__global__ void k_fft_l(const float* __restrict__ x, const _Float16* __restrict__ TL,
                        float* __restrict__ S1r, float* __restrict__ S1i){
  int bid = blockIdx.x;                 // 16384 spatial * 8 c-tiles
  int ct = bid & 7; int sp = bid >> 3;  // sp = (b*64+h)*64+w
  int lane = threadIdx.x;
  int n = lane & 15, hi8 = (lane >> 4) * 8;
  int c = ct * 16 + n;
  v16h a = ((const v16h*)TL)[lane];
  const float* xb = x + (size_t)sp * 4096 + c;     // l stride = 128
  v16h bb;
  #pragma unroll
  for (int h = 0; h < 16; h++) bb[h] = (_Float16)xb[(size_t)kmap(h, hi8) * 128];
  v8f acc = {};
  acc = WMMA(a, bb, acc);
  // D: lanes<16 -> real t=j ; lanes>=16 -> imag t=j
  float* outp = (lane < 16) ? S1r : S1i;
  size_t ob = (size_t)sp * 1024 + c;               // t stride = 128
  #pragma unroll
  for (int j = 0; j < 8; j++) outp[ob + (size_t)j * 128] = acc[j];
}

// ---------------- Stage 2: DFT over W (64 -> 32 modes), complex via K-cat=128 --------------
__global__ void k_dft_w(const float* __restrict__ S1r_, const _Float16* __restrict__ TW,
                        float* __restrict__ S2r, float* __restrict__ S2i){
  int bid = blockIdx.x;                 // 4*64*8*8 = 16384
  int ct = bid & 7, t = (bid >> 3) & 7, h = (bid >> 6) & 63, b = bid >> 12;
  int lane = threadIdx.x; int n = lane & 15; int hi8 = (lane >> 4) * 8;
  int c = ct * 16 + n;
  const float* S1i_ = S1r_ + SZ_S1;
  size_t base = (size_t)(b * 64 + h) * 65536 + (size_t)t * 128 + c;  // w stride = 1024
  v16h Bp[4];
  #pragma unroll
  for (int ch = 0; ch < 4; ch++) {
    const float* src = (ch < 2) ? S1r_ : S1i_;
    #pragma unroll
    for (int hh = 0; hh < 16; hh++) {
      int w = (ch & 1) * 32 + kmap(hh, hi8);
      Bp[ch][hh] = (_Float16)src[base + (size_t)w * 1024];
    }
  }
  const v16h* Tv = (const v16h*)TW;
  #pragma unroll
  for (int kwt = 0; kwt < 2; kwt++) {
    #pragma unroll
    for (int mat = 0; mat < 2; mat++) {
      v8f acc = {};
      #pragma unroll
      for (int ch = 0; ch < 4; ch++) {
        v16h a = Tv[((kwt * 2 + mat) * 4 + ch) * 32 + lane];
        acc = WMMA(a, Bp[ch], acc);
      }
      float* outp = mat ? S2i : S2r;
      #pragma unroll
      for (int j = 0; j < 8; j++) {
        int kw = kwt * 16 + j + hi8;
        outp[((size_t)((b * 64 + h) * 32 + kw) * 8 + t) * 128 + c] = acc[j];
      }
    }
  }
}

// ---------------- Stage 3: DFT over H (64 -> 32 modes) ------------------------------------
__global__ void k_dft_h(const float* __restrict__ S2r, const _Float16* __restrict__ TW,
                        float* __restrict__ Xr, float* __restrict__ Xi){
  int bid = blockIdx.x;                 // 4*32*8*8 = 8192
  int ct = bid & 7, t = (bid >> 3) & 7, kw = (bid >> 6) & 31, b = bid >> 11;
  int lane = threadIdx.x; int n = lane & 15; int hi8 = (lane >> 4) * 8;
  int c = ct * 16 + n;
  const float* S2i = S2r + SZ_S2;
  size_t base = (size_t)b * 2097152 + (size_t)kw * 1024 + (size_t)t * 128 + c;  // h stride = 32768
  v16h Bp[4];
  #pragma unroll
  for (int ch = 0; ch < 4; ch++) {
    const float* src = (ch < 2) ? S2r : S2i;
    #pragma unroll
    for (int hh = 0; hh < 16; hh++) {
      int hsp = (ch & 1) * 32 + kmap(hh, hi8);
      Bp[ch][hh] = (_Float16)src[base + (size_t)hsp * 32768];
    }
  }
  const v16h* Tv = (const v16h*)TW;
  #pragma unroll
  for (int kht = 0; kht < 2; kht++) {
    #pragma unroll
    for (int mat = 0; mat < 2; mat++) {
      v8f acc = {};
      #pragma unroll
      for (int ch = 0; ch < 4; ch++) {
        v16h a = Tv[((kht * 2 + mat) * 4 + ch) * 32 + lane];
        acc = WMMA(a, Bp[ch], acc);
      }
      float* outp = mat ? Xi : Xr;
      #pragma unroll
      for (int j = 0; j < 8; j++) {
        int kh = kht * 16 + j + hi8;
        outp[((size_t)((b * 32 + kh) * 32 + kw) * 8 + t) * 128 + c] = acc[j];
      }
    }
  }
}

// ---------------- MLP: per block nb, complex 16x16 layers with fused LoRA, exact GELU ------
__global__ void k_mlp(const float* __restrict__ Xr, const _Float16* __restrict__ WPK,
                      const float* __restrict__ b1, const float* __restrict__ b2,
                      float* __restrict__ Or, float* __restrict__ Oi_unused){
  __shared__ float lds[2][16][17];
  int t0 = blockIdx.x * 16;             // 32768 tokens / 16
  int lane = threadIdx.x; int n = lane & 15; int hi = lane >> 4; int hi8 = hi * 8;
  const float* Xi = Xr + SZ_S3;
  float* Oi = Or + SZ_S3;
  const v16h* WP = (const v16h*)WPK;
  int tok = t0 + n;
  for (int nb = 0; nb < 8; nb++) {
    v16h Bc;
    const float* xr = Xr + (size_t)tok * 128 + nb * 16 + hi8;
    const float* xi = Xi + (size_t)tok * 128 + nb * 16 + hi8;
    #pragma unroll
    for (int h = 0; h < 8; h++) { Bc[h] = (_Float16)xr[h]; Bc[h + 8] = (_Float16)xi[h]; }
    v8f cd, cs;
    const float* b1r = b1 + nb * 16 + hi8;
    const float* b1i = b1 + 128 + nb * 16 + hi8;
    #pragma unroll
    for (int j = 0; j < 8; j++) { float br = b1r[j], bi = b1i[j]; cd[j] = br - bi; cs[j] = br + bi; }
    v16h a1r = WP[(nb * 4 + 0) * 32 + lane];
    v16h a1i = WP[(nb * 4 + 1) * 32 + lane];
    v8f ur = WMMA(a1r, Bc, cd);   // xr@W1r - xi@W1i + (b1r-b1i)  (transposed orientation)
    v8f ui = WMMA(a1i, Bc, cs);   // xr@W1i + xi@W1r + (b1r+b1i)
    #pragma unroll
    for (int j = 0; j < 8; j++) {
      ur[j] = 0.5f * ur[j] * (1.f + erff(ur[j] * 0.70710678118654752f));
      ui[j] = 0.5f * ui[j] * (1.f + erff(ui[j] * 0.70710678118654752f));
    }
    // layer-2 B comes straight from D registers (in-lane, no transpose)
    v16h Bc2;
    #pragma unroll
    for (int j = 0; j < 8; j++) { Bc2[j] = (_Float16)ur[j]; Bc2[j + 8] = (_Float16)ui[j]; }
    v8f cd2, cs2;
    const float* b2r = b2 + nb * 16 + hi8;
    const float* b2i = b2 + 128 + nb * 16 + hi8;
    #pragma unroll
    for (int j = 0; j < 8; j++) { float br = b2r[j], bi = b2i[j]; cd2[j] = br - bi; cs2[j] = br + bi; }
    v16h a2r = WP[(nb * 4 + 2) * 32 + lane];
    v16h a2i = WP[(nb * 4 + 3) * 32 + lane];
    v8f o2r = WMMA(a2r, Bc2, cd2);
    v8f o2i = WMMA(a2i, Bc2, cs2);
    // transpose via LDS for coalesced 64B-per-token stores
    __syncthreads();
    #pragma unroll
    for (int j = 0; j < 8; j++) { lds[0][n][j + hi8] = o2r[j]; lds[1][n][j + hi8] = o2i[j]; }
    __syncthreads();
    float* dst = (hi ? Oi : Or) + (size_t)(t0 + n) * 128 + nb * 16;
    #pragma unroll
    for (int f = 0; f < 16; f++) dst[f] = lds[hi][n][f];
  }
  (void)Oi_unused;
}

// ---------------- Stage 6: inverse DFT over H (32 -> 64), K-cat=64 -------------------------
__global__ void k_idft_h(const float* __restrict__ Or, const _Float16* __restrict__ TIW,
                         float* __restrict__ Hr, float* __restrict__ Hi){
  int bid = blockIdx.x;                 // 8192: (b,kw,t,ct)
  int ct = bid & 7, t = (bid >> 3) & 7, kw = (bid >> 6) & 31, b = bid >> 11;
  int lane = threadIdx.x; int n = lane & 15; int hi8 = (lane >> 4) * 8;
  int c = ct * 16 + n;
  const float* Oi = Or + SZ_S3;
  size_t base = (size_t)b * 1048576 + (size_t)kw * 1024 + (size_t)t * 128 + c;  // kh stride = 32768
  v16h Bp[2];
  #pragma unroll
  for (int ch = 0; ch < 2; ch++) {
    const float* src = ch ? Oi : Or;
    #pragma unroll
    for (int hh = 0; hh < 16; hh++)
      Bp[ch][hh] = (_Float16)src[base + (size_t)kmap(hh, hi8) * 32768];
  }
  const v16h* Tv = (const v16h*)TIW;
  #pragma unroll
  for (int mt = 0; mt < 4; mt++) {
    #pragma unroll
    for (int mat = 0; mat < 2; mat++) {
      v8f acc = {};
      #pragma unroll
      for (int ch = 0; ch < 2; ch++) {
        v16h a = Tv[((mt * 2 + mat) * 2 + ch) * 32 + lane];
        acc = WMMA(a, Bp[ch], acc);
      }
      float* outp = mat ? Hi : Hr;
      #pragma unroll
      for (int j = 0; j < 8; j++) {
        int hh_ = mt * 16 + j + hi8;
        outp[((size_t)((b * 64 + hh_) * 32 + kw) * 8 + t) * 128 + c] = acc[j];
      }
    }
  }
}

// ---------------- Stage 7: inverse DFT over W (32 -> 64) ----------------------------------
__global__ void k_idft_w(const float* __restrict__ Hr, const _Float16* __restrict__ TIW,
                         float* __restrict__ Wr, float* __restrict__ Wi){
  int bid = blockIdx.x;                 // 16384: (b,h,t,ct)
  int ct = bid & 7, t = (bid >> 3) & 7, h = (bid >> 6) & 63, b = bid >> 12;
  int lane = threadIdx.x; int n = lane & 15; int hi8 = (lane >> 4) * 8;
  int c = ct * 16 + n;
  const float* Hi = Hr + SZ_S2;
  size_t base = (size_t)(b * 64 + h) * 32768 + (size_t)t * 128 + c;  // kw stride = 1024
  v16h Bp[2];
  #pragma unroll
  for (int ch = 0; ch < 2; ch++) {
    const float* src = ch ? Hi : Hr;
    #pragma unroll
    for (int hh = 0; hh < 16; hh++)
      Bp[ch][hh] = (_Float16)src[base + (size_t)kmap(hh, hi8) * 1024];
  }
  const v16h* Tv = (const v16h*)TIW;
  #pragma unroll
  for (int mt = 0; mt < 4; mt++) {
    #pragma unroll
    for (int mat = 0; mat < 2; mat++) {
      v8f acc = {};
      #pragma unroll
      for (int ch = 0; ch < 2; ch++) {
        v16h a = Tv[((mt * 2 + mat) * 2 + ch) * 32 + lane];
        acc = WMMA(a, Bp[ch], acc);
      }
      float* outp = mat ? Wi : Wr;
      #pragma unroll
      for (int j = 0; j < 8; j++) {
        int w = mt * 16 + j + hi8;
        outp[((size_t)((b * 64 + h) * 64 + w) * 8 + t) * 128 + c] = acc[j];
      }
    }
  }
}

// ---------------- Stage 8: inverse rfft along L (8 modes -> 32) + residual ----------------
__global__ void k_ifft_l(const float* __restrict__ Wr_, const float* __restrict__ x,
                         const _Float16* __restrict__ TIL, float* __restrict__ out){
  int bid = blockIdx.x;                 // 16384 spatial * 8 c-tiles
  int ct = bid & 7; int sp = bid >> 3;
  int lane = threadIdx.x; int n = lane & 15; int hi8 = (lane >> 4) * 8;
  int c = ct * 16 + n;
  const float* Wi_ = Wr_ + SZ_S1;
  size_t baseF = (size_t)sp * 1024 + c;  // t stride = 128
  // Kcat=16: lanes<16 carry Fr k=0..7 (k-set 0-7), lanes>=16 carry Fi k=0..7 (k-set 8-15)
  const float* srcp = (lane < 16) ? Wr_ : Wi_;
  v16h Bp;
  #pragma unroll
  for (int hh = 0; hh < 8; hh++) {
    Bp[hh]     = (_Float16)srcp[baseF + (size_t)hh * 128];
    Bp[hh + 8] = (_Float16)0.f;
  }
  const v16h* Tv = (const v16h*)TIL;
  size_t xbase = (size_t)sp * 4096 + c;
  #pragma unroll
  for (int mt = 0; mt < 2; mt++) {
    v16h a = Tv[mt * 32 + lane];
    v8f acc = {};
    acc = WMMA(a, Bp, acc);
    #pragma unroll
    for (int j = 0; j < 8; j++) {
      int li = mt * 16 + j + hi8;
      size_t idx = xbase + (size_t)li * 128;
      out[idx] = acc[j] + x[idx];
    }
  }
}

// ---------------- Host orchestration -------------------------------------------------------
extern "C" void kernel_launch(void* const* d_in, const int* in_sizes, int n_in,
                              void* d_out, int out_size, void* d_ws, size_t ws_size,
                              hipStream_t stream) {
  (void)in_sizes; (void)n_in; (void)out_size; (void)ws_size;
  const float* x   = (const float*)d_in[0];
  const float* w1  = (const float*)d_in[1];
  const float* b1  = (const float*)d_in[2];
  const float* w2  = (const float*)d_in[3];
  const float* b2  = (const float*)d_in[4];
  const float* A1r = (const float*)d_in[5];
  const float* B1r = (const float*)d_in[6];
  const float* A1i = (const float*)d_in[7];
  const float* B1i = (const float*)d_in[8];
  const float* A2r = (const float*)d_in[9];
  const float* B2r = (const float*)d_in[10];
  const float* A2i = (const float*)d_in[11];
  const float* B2i = (const float*)d_in[12];
  const float* ew  = (const float*)d_in[13];
  float* out = (float*)d_out;

  // ws layout (floats): A0[2*SZ_S1] | B0[2*SZ_S2] | tables (halves) — total ~192.1 MiB
  float* A0 = (float*)d_ws;
  float* B0 = A0 + 2 * SZ_S1;
  _Float16* TAB = (_Float16*)(B0 + 2 * SZ_S2);
  _Float16* TL  = TAB;                              // 512 halves
  _Float16* TW  = TAB + 512;                        // 16*512
  _Float16* TIW = TAB + 512 + 8192;                 // 16*512
  _Float16* TIL = TAB + 512 + 8192 + 8192;          // 2*512
  _Float16* WPK = TAB + 512 + 8192 + 8192 + 1024;   // 8*4*512

  k_init_tables<<<35, 512, 0, stream>>>(TAB);
  k_init_weff<<<8, 256, 0, stream>>>(w1, b1, w2, b2, A1r, B1r, A1i, B1i,
                                     A2r, B2r, A2i, B2i, ew, WPK);
  // S1 (r,i) -> A0
  k_fft_l<<<131072, 32, 0, stream>>>(x, TL, A0, A0 + SZ_S1);
  // S2 (r,i) -> B0
  k_dft_w<<<16384, 32, 0, stream>>>(A0, TW, B0, B0 + SZ_S2);
  // kept modes X (r,i) -> A0[0 .. 2*SZ_S3)
  k_dft_h<<<8192, 32, 0, stream>>>(B0, TW, A0, A0 + SZ_S3);
  // MLP out O (r,i) -> A0[2*SZ_S3 .. 4*SZ_S3)
  k_mlp<<<2048, 32, 0, stream>>>(A0, WPK, b1, b2, A0 + 2 * SZ_S3, A0 + 3 * SZ_S3);
  // inverse H -> B0
  k_idft_h<<<8192, 32, 0, stream>>>(A0 + 2 * SZ_S3, TIW, B0, B0 + SZ_S2);
  // inverse W -> A0
  k_idft_w<<<16384, 32, 0, stream>>>(B0, TIW, A0, A0 + SZ_S1);
  // inverse L + residual -> out
  k_ifft_l<<<131072, 32, 0, stream>>>(A0, x, TIL, out);
}
